// PointNetGrid_87101936763208
// MI455X (gfx1250) — matrix operand order repaired and checked by reading, not compiled, not run on hardware
//
#include <hip/hip_runtime.h>
#include <hip/hip_bf16.h>
#include <stdint.h>

// ---------------------------------------------------------------------------
// CDNA5 (gfx1250) PointNet++ grid pipeline.
// GEMMs: v_wmma_f32_16x16x32_bf16 (wave32), A pre-packed in fragment order,
// B pre-packed bf16 tiles async-copied to LDS (global_load_async_to_lds_b64,
// ASYNCcnt double buffering), branch-free steady-state loop.
// ---------------------------------------------------------------------------

typedef __attribute__((ext_vector_type(16))) __bf16 v16bf;
typedef __attribute__((ext_vector_type(8)))  float  v8f;

// ---------------------------------------------------------------------------
// Pack W (CO x CI f32) -> bf16 A fragments, zero-padded to COp x CIp:
// Apack[((cot*nkt + kt)*32 + lane)*16 + e]
// A layout: m = lane&15, grp = lane>>4; e -> v=e>>1, h=e&1,
//           k = (v&4 ? 16:0) + (v&3)*2 + h + grp*8.
// ---------------------------------------------------------------------------
__global__ void pack_w_kernel(const float* __restrict__ W, __bf16* __restrict__ Apack,
                              int CO, int CI, int COp, int CIp)
{
    const int ncot = COp >> 4;
    const int nkt  = CIp >> 5;
    long long total = (long long)ncot * nkt * 512;
    long long i = (long long)blockIdx.x * blockDim.x + threadIdx.x;
    if (i >= total) return;
    int e    = (int)(i & 15);
    int lane = (int)((i >> 4) & 31);
    long long tile = i >> 9;
    int kt  = (int)(tile % nkt);
    int cot = (int)(tile / nkt);
    int grp = lane >> 4;
    int co  = cot * 16 + (lane & 15);
    int v = e >> 1, h = e & 1;
    int k = kt * 32 + ((v & 4) ? 16 : 0) + ((v & 3) * 2) + h + grp * 8;
    float val = (co < CO && k < CI) ? W[(size_t)co * (size_t)CI + k] : 0.f;
    Apack[i] = (__bf16)val;
}

// ---------------------------------------------------------------------------
// Pack X (CIp x M f32, pad rows zero) -> bf16 B tiles in fragment order:
// Bpack[((mt*nkt + kt)*1024) + n*32 + p]  (n,p in [0,32)), k = ((p&8)<<1)|((p>>4)<<3)|(p&7)
// value = X[(kt*32 + k)*M + mt*32 + n].
// ---------------------------------------------------------------------------
__global__ void pack_b_kernel(const float* __restrict__ X, __bf16* __restrict__ Bpack,
                              int CIp, int M)
{
    const int nkt = CIp >> 5;
    long long total = (long long)(M >> 5) * nkt * 1024;
    long long i = (long long)blockIdx.x * blockDim.x + threadIdx.x;
    if (i >= total) return;
    int p = (int)(i & 31);
    int n = (int)((i >> 5) & 31);
    int k = ((p & 8) << 1) | ((p >> 4) << 3) | (p & 7);
    long long tile = i >> 10;
    int kt = (int)(tile % nkt);
    int mt = (int)(tile / nkt);
    Bpack[i] = (__bf16)X[(size_t)(kt * 32 + k) * (size_t)M + (size_t)(mt * 32 + n)];
}

// ---------------------------------------------------------------------------
// GEMM: Y[co, m] = act( sum W*X + bias ).  M mult 32, CIp mult 32, COp mult 32.
// Block = 256 thr = 8 waves; tile = 128(CO) x 32(M).
// B tiles (2 KB) async-copied to LDS, double-buffered via ASYNCcnt:
// prologue issue -> steady loop {issue next; wait<=1; barrier; 2x WMMA;
// barrier} -> peeled last step {wait 0; barrier; 2x WMMA}.
// ---------------------------------------------------------------------------
__global__ __launch_bounds__(256)
void gemm_wmma(const __bf16* __restrict__ Apack,
               const __bf16* __restrict__ Bpack,
               const float* __restrict__ bias,
               float* __restrict__ Y,
               int CO, int COp, int CIp, int M, int do_relu)
{
    __shared__ __align__(32) __bf16 sB[2][1024];   // two 2 KB B-tile buffers

    const int tid   = threadIdx.x;
    const int lane  = tid & 31;
    const int grp   = lane >> 4;
    const int lan16 = lane & 15;

    const int mt   = blockIdx.x;
    const int m0   = mt * 32;
    const int cot  = blockIdx.y * 8 + (tid >> 5);
    const int ncot = COp >> 4;
    const int nkt  = CIp >> 5;

    v8f acc0 = {0.f, 0.f, 0.f, 0.f, 0.f, 0.f, 0.f, 0.f};
    v8f acc1 = acc0;

    // Clamp out-of-range CO tiles to the last valid one: they compute garbage
    // but never store (epilogue guard), keeping the hot loop predication-free.
    const int cotc = (cot < ncot) ? cot : (ncot - 1);
    const __bf16* aptr = Apack + ((size_t)cotc * (size_t)nkt) * 512 + (size_t)lane * 16;
    const __bf16* bsrc = Bpack + ((size_t)mt * (size_t)nkt) * 1024 + (size_t)tid * 4;

    const unsigned lds0 = (unsigned)(uintptr_t)&sB[0][tid * 4];
    const unsigned lds1 = (unsigned)(uintptr_t)&sB[1][tid * 4];

    auto wmma_step = [&](int kt) {
        const v16bf a = *(const v16bf*)(aptr + (size_t)kt * 512);
        const __bf16* bt = &sB[kt & 1][0];
        const v16bf b0 = *(const v16bf*)(bt + lan16 * 32 + grp * 16);
        const v16bf b1 = *(const v16bf*)(bt + (16 + lan16) * 32 + grp * 16);
        acc0 = __builtin_amdgcn_wmma_f32_16x16x32_bf16(false, a, false, b0,
                                                       (short)0, acc0, false, false);
        acc1 = __builtin_amdgcn_wmma_f32_16x16x32_bf16(false, a, false, b1,
                                                       (short)0, acc1, false, false);
    };

    // prologue: tile 0 -> buffer 0
    asm volatile("global_load_async_to_lds_b64 %0, %1, off"
                 :: "v"(lds0),
                    "v"((unsigned long long)(uintptr_t)bsrc)
                 : "memory");

    // steady state: branch-free body
    for (int kt = 0; kt < nkt - 1; ++kt) {
        const unsigned ldsn = (kt & 1) ? lds0 : lds1;   // next buffer
        asm volatile("global_load_async_to_lds_b64 %0, %1, off"
                     :: "v"(ldsn),
                        "v"((unsigned long long)(uintptr_t)(bsrc + (size_t)(kt + 1) * 1024))
                     : "memory");
        asm volatile("s_wait_asynccnt 0x1" ::: "memory");   // tile kt complete
        __syncthreads();
        wmma_step(kt);
        __syncthreads();
    }

    // peeled last step
    asm volatile("s_wait_asynccnt 0x0" ::: "memory");
    __syncthreads();
    wmma_step(nkt - 1);

    // epilogue: D layout -> col = lane&15 (+16 for acc1), row r -> co + r + grp*8
    const int co_base = cot * 16;
    if (co_base < COp) {
        #pragma unroll
        for (int r = 0; r < 8; ++r) {
            const int co = co_base + r + grp * 8;
            float v0 = 0.f, v1 = 0.f;
            if (co < CO) {
                const float bb = bias[co];
                v0 = acc0[r] + bb;
                v1 = acc1[r] + bb;
                if (do_relu) { v0 = fmaxf(v0, 0.f); v1 = fmaxf(v1, 0.f); }
            }
            Y[(size_t)co * (size_t)M + m0 + lan16]      = v0;
            Y[(size_t)co * (size_t)M + m0 + 16 + lan16] = v1;
        }
    }
}

// ---------------------------------------------------------------------------
// Zero rows [rowBegin, rowEnd) of an (rows x M) f32 matrix.
// ---------------------------------------------------------------------------
__global__ void zero_rows_kernel(float* __restrict__ X, int rowBegin, int rowEnd, int M)
{
    long long total = (long long)(rowEnd - rowBegin) * (long long)M;
    long long i = (long long)blockIdx.x * blockDim.x + threadIdx.x;
    if (i >= total) return;
    X[(size_t)rowBegin * (size_t)M + i] = 0.f;
}

// ---------------------------------------------------------------------------
// Furthest point sampling (B=1), pointnet2 semantics.
// ---------------------------------------------------------------------------
__global__ __launch_bounds__(1024)
void fps_kernel(const float* __restrict__ xyz, int N, int npoint, int* __restrict__ idx)
{
    __shared__ float mind[2048];
    __shared__ float smax[1024];
    __shared__ int   simax[1024];
    __shared__ int   s_last;

    int t = threadIdx.x;
    for (int i = t; i < N; i += 1024) mind[i] = 1e10f;
    if (t == 0) { s_last = 0; idx[0] = 0; }
    __syncthreads();

    for (int s = 1; s < npoint; ++s) {
        int last = s_last;
        float lx = xyz[last * 3 + 0];
        float ly = xyz[last * 3 + 1];
        float lz = xyz[last * 3 + 2];
        float best = -1e30f; int bi = 0;
        for (int i = t; i < N; i += 1024) {
            float dx = xyz[i * 3 + 0] - lx;
            float dy = xyz[i * 3 + 1] - ly;
            float dz = xyz[i * 3 + 2] - lz;
            float d  = dx * dx + dy * dy + dz * dz;
            float mn = fminf(mind[i], d);
            mind[i] = mn;
            if (mn > best) { best = mn; bi = i; }
        }
        smax[t] = best; simax[t] = bi;
        __syncthreads();
        for (int str = 512; str > 0; str >>= 1) {
            if (t < str) {
                float o = smax[t + str]; int oi = simax[t + str];
                if (o > smax[t] || (o == smax[t] && oi < simax[t])) {
                    smax[t] = o; simax[t] = oi;
                }
            }
            __syncthreads();
        }
        if (t == 0) { s_last = simax[0]; idx[s] = simax[0]; }
        __syncthreads();
    }
}

__global__ void gather_centers_kernel(const float* __restrict__ xyz,
                                      const int* __restrict__ idx,
                                      int S, float* __restrict__ nxyz)
{
    int s = blockIdx.x * blockDim.x + threadIdx.x;
    if (s >= S) return;
    int g = idx[s];
    nxyz[s * 3 + 0] = xyz[g * 3 + 0];
    nxyz[s * 3 + 1] = xyz[g * 3 + 1];
    nxyz[s * 3 + 2] = xyz[g * 3 + 2];
}

// ---------------------------------------------------------------------------
// Ball query: first nsample in-radius points (index order), pad with first.
// ---------------------------------------------------------------------------
__global__ void ball_query_kernel(float r2, int ns,
                                  const float* __restrict__ xyz, int N,
                                  const float* __restrict__ new_xyz, int S,
                                  int* __restrict__ out)
{
    int s = blockIdx.x * blockDim.x + threadIdx.x;
    if (s >= S) return;
    float cx = new_xyz[s * 3 + 0];
    float cy = new_xyz[s * 3 + 1];
    float cz = new_xyz[s * 3 + 2];
    int cnt = 0, first = 0;
    for (int i = 0; i < N && cnt < ns; ++i) {
        float dx = xyz[i * 3 + 0] - cx;
        float dy = xyz[i * 3 + 1] - cy;
        float dz = xyz[i * 3 + 2] - cz;
        float d2 = dx * dx + dy * dy + dz * dz;
        if (d2 < r2) {
            if (cnt == 0) first = i;
            out[s * ns + cnt] = i;
            ++cnt;
        }
    }
    for (; cnt < ns; ++cnt) out[s * ns + cnt] = first;
}

// ---------------------------------------------------------------------------
// Grouped input X (rows 0..2: centered xyz; rows 3..3+C-1: features).
// ---------------------------------------------------------------------------
__global__ void gather_group_kernel(const float* __restrict__ xyz,
                                    const float* __restrict__ feats,  // C x N (may be null)
                                    const float* __restrict__ new_xyz,
                                    const int* __restrict__ gi,
                                    int S, int ns, int C, int N,
                                    float* __restrict__ X)
{
    int col = blockIdx.x * blockDim.x + threadIdx.x;
    size_t M = (size_t)S * (size_t)ns;
    if (col >= (int)M) return;
    int s = col / ns;
    int g = gi[col];
    X[0 * M + col] = xyz[g * 3 + 0] - new_xyz[s * 3 + 0];
    X[1 * M + col] = xyz[g * 3 + 1] - new_xyz[s * 3 + 1];
    X[2 * M + col] = xyz[g * 3 + 2] - new_xyz[s * 3 + 2];
    for (int c = 0; c < C; ++c)
        X[(size_t)(3 + c) * M + col] = feats[(size_t)c * (size_t)N + g];
}

// Max over nsample -> F[(row_off+c)*ldF + s]
__global__ void maxpool_kernel(const float* __restrict__ Y, int C, int S, int ns,
                               float* __restrict__ F, int row_off, int ldF)
{
    int i = blockIdx.x * blockDim.x + threadIdx.x;
    if (i >= C * S) return;
    int c = i / S, s = i % S;
    size_t M = (size_t)S * (size_t)ns;
    const float* p = Y + (size_t)c * M + (size_t)s * ns;
    float m = -1e30f;
    for (int j = 0; j < ns; ++j) m = fmaxf(m, p[j]);
    F[(size_t)(row_off + c) * (size_t)ldF + s] = m;
}

// ---------------------------------------------------------------------------
// Voxel prep (V = 1, center (-1,-1,-1), half 0.5)
// ---------------------------------------------------------------------------
__global__ void voxel_member_kernel(const float* __restrict__ scene, int N,
                                    float cx, float cy, float cz, float half,
                                    int* __restrict__ member, int* __restrict__ count)
{
    if (blockIdx.x != 0 || threadIdx.x != 0) return;
    int c = 0;
    for (int i = 0; i < N; ++i) {
        if (fabsf(scene[i * 3 + 0] - cx) < half &&
            fabsf(scene[i * 3 + 1] - cy) < half &&
            fabsf(scene[i * 3 + 2] - cz) < half)
            member[c++] = i;
    }
    *count = c;
}

__global__ void sample_scene_kernel(const float* __restrict__ scene,
                                    const int* __restrict__ member,
                                    const int* __restrict__ count,
                                    const int* __restrict__ rinds,
                                    float cx, float cy, float cz,
                                    float* __restrict__ norm_scene)
{
    int j = blockIdx.x * blockDim.x + threadIdx.x;
    if (j >= 1024) return;
    int c = *count;
    int s = (c > 0) ? (rinds[j] % c) : 0;
    int g = member[s];
    norm_scene[j * 3 + 0] = scene[g * 3 + 0] - cx;
    norm_scene[j * 3 + 1] = scene[g * 3 + 1] - cy;
    norm_scene[j * 3 + 2] = scene[g * 3 + 2] - cz;
}

__global__ void tr_select_kernel(const float* __restrict__ tr, int T,
                                 float cx, float cy, float cz, float half,
                                 int* __restrict__ tv)
{
    if (blockIdx.x != 0 || threadIdx.x != 0) return;
    int c = 0;
    for (int t = 0; t < T; ++t) {
        if (fabsf(tr[t * 3 + 0] - cx) < half &&
            fabsf(tr[t * 3 + 1] - cy) < half &&
            fabsf(tr[t * 3 + 2] - cz) < half)
            tv[c++] = t;
    }
    for (int i = c; i < T; ++i) tv[i] = 0;
}

// ---------------------------------------------------------------------------
// Classifier input X (Cp x T)
// ---------------------------------------------------------------------------
__global__ void build_cls_in_kernel(const float* __restrict__ objf,
                                    const float* __restrict__ scnf,
                                    const float* __restrict__ tr,
                                    const float* __restrict__ rot,
                                    const int* __restrict__ tv,
                                    float cx, float cy, float cz,
                                    int T, int Cp, float* __restrict__ X)
{
    long long i = (long long)blockIdx.x * blockDim.x + threadIdx.x;
    if (i >= (long long)Cp * T) return;
    int row = (int)(i / T);
    int t   = (int)(i % T);
    float v = 0.f;
    if (row < 1024) {
        v = objf[row];
    } else if (row < 5120) {
        v = scnf[row - 1024];
    } else if (row < 5123) {
        int tt = tv[t];
        int d = row - 5120;
        float cc = (d == 0) ? cx : (d == 1) ? cy : cz;
        v = tr[tt * 3 + d] - cc;
    } else if (row < 5129) {
        int tt = tv[t];
        v = rot[tt * 6 + (row - 5123)];
    }
    X[(size_t)row * (size_t)T + t] = v;
}

__global__ void zero_out_kernel(float* __restrict__ out, int n)
{
    int i = blockIdx.x * blockDim.x + threadIdx.x;
    if (i < n) out[i] = 0.f;
}

// out = zeros(T).at[tv].add(pred) / 8
__global__ void scatter_kernel(const float* __restrict__ pred,
                               const int* __restrict__ tv,
                               int T, float* __restrict__ out)
{
    int t = blockIdx.x * blockDim.x + threadIdx.x;
    if (t >= T) return;
    atomicAdd(&out[tv[t]], pred[t] * 0.125f);
}

// ---------------------------------------------------------------------------
// Host orchestration
// ---------------------------------------------------------------------------
static inline int cdiv(int a, int b) { return (a + b - 1) / b; }
static inline int rup(int a, int b)  { return ((a + b - 1) / b) * b; }

extern "C" void kernel_launch(void* const* d_in, const int* in_sizes, int n_in,
                              void* d_out, int out_size, void* d_ws, size_t ws_size,
                              hipStream_t stream)
{
    (void)in_sizes; (void)n_in; (void)out_size; (void)ws_size;

    const float* scene = (const float*)d_in[0];   // (1,131072,3)
    const float* obj   = (const float*)d_in[1];   // (1,2048,3)
    const float* tr    = (const float*)d_in[2];   // (4096,3)
    const float* rot   = (const float*)d_in[3];   // (4096,6)
    const int*   rinds = (const int*)d_in[4];     // (1024,1)
    int pi = 5;                                   // params: obj, scene, cls (pytree order)

    const float CX = -1.f, CY = -1.f, CZ = -1.f, HALF = 0.5f;
    const int T = 4096;

    // ---- workspace bump allocator ----
    char* wbase = (char*)d_ws;
    size_t off = 0;
    auto alloc = [&](size_t bytes) -> void* {
        void* p = wbase + off;
        off = (off + bytes + 255) & ~(size_t)255;
        return p;
    };
    int*    member     = (int*)alloc((size_t)131072 * 4);
    int*    vcount     = (int*)alloc(256);
    float*  norm_scene = (float*)alloc((size_t)1024 * 3 * 4);
    int*    tv         = (int*)alloc((size_t)T * 4);
    int*    fps_idx    = (int*)alloc((size_t)8192 * 4);
    float*  nxyzA      = (float*)alloc((size_t)8192 * 3 * 4);
    float*  nxyzB      = (float*)alloc((size_t)8192 * 3 * 4);
    int*    gi         = (int*)alloc((size_t)262144 * 4);
    float*  featA      = (float*)alloc((size_t)8 * 1024 * 1024);
    float*  featB      = (float*)alloc((size_t)8 * 1024 * 1024);
    float*  objfeat    = (float*)alloc((size_t)1024 * 4);
    __bf16* bigW       = (__bf16*)alloc((size_t)32 * 1024 * 1024);
    __bf16* bigB       = (__bf16*)alloc((size_t)48 * 1024 * 1024);
    float*  bigX       = (float*)alloc((size_t)96 * 1024 * 1024);
    float*  bigY0      = (float*)alloc((size_t)80 * 1024 * 1024);
    float*  bigY1      = (float*)alloc((size_t)80 * 1024 * 1024);

    // pack W + pack B, then run the WMMA GEMM (CIp/M multiples of 32)
    auto run_gemm = [&](const float* W, const float* Xc, const float* bb, float* Yo,
                        int CO, int CI, int CIp, int M, int relu) {
        int COp = rup(CO, 32);
        long long pw = (long long)(COp >> 4) * (CIp >> 5) * 512;
        pack_w_kernel<<<cdiv((int)pw, 256), 256, 0, stream>>>(W, bigW, CO, CI, COp, CIp);
        long long pb = (long long)(M >> 5) * (CIp >> 5) * 1024;
        pack_b_kernel<<<cdiv((int)pb, 256), 256, 0, stream>>>(Xc, bigB, CIp, M);
        dim3 g(M / 32, cdiv(COp, 128));
        gemm_wmma<<<g, 256, 0, stream>>>(bigW, bigB, bb, Yo, CO, COp, CIp, M, relu);
        return COp;
    };

    // ---- voxel prep ----
    voxel_member_kernel<<<1, 1, 0, stream>>>(scene, 131072, CX, CY, CZ, HALF, member, vcount);
    sample_scene_kernel<<<cdiv(1024, 256), 256, 0, stream>>>(scene, member, vcount, rinds,
                                                             CX, CY, CZ, norm_scene);
    tr_select_kernel<<<1, 1, 0, stream>>>(tr, T, CX, CY, CZ, HALF, tv);

    // ---- tower configs ----
    static const int scene_np[6] = {8192, 2048, 512, 128, 32, 1};
    static const float scene_r[6][2] = {{0.005f,0.01f},{0.01f,0.05f},{0.1f,0.2f},
                                        {0.2f,0.4f},{0.4f,0.8f},{0.8f,1.6f}};
    static const int scene_mlps[6][2][3] = {
        {{16,16,32},{32,32,64}}, {{64,64,128},{64,96,128}},
        {{128,196,256},{128,196,256}}, {{256,256,512},{256,384,512}},
        {{512,768,1024},{512,768,1024}}, {{1024,1024,2048},{1024,1536,2048}}};
    static const int obj_np[4] = {512, 256, 128, 1};
    static const float obj_r[4][2] = {{0.005f,0.01f},{0.01f,0.02f},
                                      {0.02f,0.04f},{0.04f,0.08f}};
    static const int obj_mlps[4][2][3] = {
        {{16,16,32},{32,32,64}}, {{64,64,128},{64,96,128}},
        {{128,196,256},{128,196,256}}, {{256,256,512},{256,384,512}}};

    auto run_tower = [&](const float* xyz0, int N0, int nlv, const int* nps,
                         const float (*radii)[2], const int (*mlps)[2][3]) -> float* {
        const float* xyz = xyz0;
        int N = N0;
        const float* feats = nullptr;
        int C = 0;
        for (int lvl = 0; lvl < nlv; ++lvl) {
            int S = nps[lvl];
            float* nxyz = (lvl & 1) ? nxyzB : nxyzA;
            fps_kernel<<<1, 1024, 0, stream>>>(xyz, N, S, fps_idx);
            gather_centers_kernel<<<cdiv(S, 256), 256, 0, stream>>>(xyz, fps_idx, S, nxyz);
            float* fout = (lvl & 1) ? featB : featA;
            int row_off = 0;
            for (int br = 0; br < 2; ++br) {
                const int ns = 32;
                float r = radii[lvl][br];
                int M = S * ns;
                ball_query_kernel<<<cdiv(S, 256), 256, 0, stream>>>(r * r, ns, xyz, N, nxyz, S, gi);
                gather_group_kernel<<<cdiv(M, 256), 256, 0, stream>>>(xyz, feats, nxyz, gi,
                                                                      S, ns, C, N, bigX);
                int ci  = C + 3;
                int cip = rup(ci, 32);
                if (cip > ci) {
                    long long pad = (long long)(cip - ci) * M;
                    zero_rows_kernel<<<cdiv((int)pad, 256), 256, 0, stream>>>(bigX, ci, cip, M);
                }
                const float* Xc = bigX;
                for (int ly = 0; ly < 3; ++ly) {
                    int co = mlps[lvl][br][ly];
                    const float* W  = (const float*)d_in[pi++];
                    const float* bb = (const float*)d_in[pi++];
                    float* Yo = (ly & 1) ? bigY1 : bigY0;
                    int cop = run_gemm(W, Xc, bb, Yo, co, ci, cip, M, 1);
                    Xc = Yo; ci = co; cip = cop;
                }
                maxpool_kernel<<<cdiv(ci * S, 256), 256, 0, stream>>>(Xc, ci, S, ns,
                                                                      fout, row_off, S);
                row_off += ci;
            }
            xyz = nxyz; N = S; feats = fout; C = row_off;
        }
        return (float*)feats;   // (C x 1) when final npoint == 1
    };

    // ---- obj tower (1024 x 1), copy out before scene tower reuses buffers ----
    float* objf_src = run_tower(obj, 2048, 4, obj_np, obj_r, obj_mlps);
    hipMemcpyAsync(objfeat, objf_src, (size_t)1024 * 4, hipMemcpyDeviceToDevice, stream);

    // ---- scene tower on the 1024 normalized voxel samples (4096 x 1) ----
    float* scnfeat = run_tower(norm_scene, 1024, 6, scene_np, scene_r, scene_mlps);

    // ---- classifier: 5129 -> 2048 -> 1024 -> 512 -> 512 -> 1 over T columns ----
    int ci = 5129, cip = rup(ci, 32);   // 5152
    build_cls_in_kernel<<<cdiv(cip * T, 256), 256, 0, stream>>>(objfeat, scnfeat, tr, rot, tv,
                                                                CX, CY, CZ, T, cip, bigX);
    const float* Xc = bigX;
    static const int cls_co[5] = {2048, 1024, 512, 512, 1};
    for (int k = 0; k < 5; ++k) {
        const float* W  = (const float*)d_in[pi++];
        const float* bb = (const float*)d_in[pi++];
        float* Yo = (k & 1) ? bigY1 : bigY0;
        int cop = run_gemm(W, Xc, bb, Yo, cls_co[k], ci, cip, T, (k < 4) ? 1 : 0);
        Xc = Yo; ci = cls_co[k]; cip = cop;
    }

    // ---- scatter: out = zeros(T).at[tv].add(pred) / 8 ----
    zero_out_kernel<<<cdiv(T, 256), 256, 0, stream>>>((float*)d_out, T);
    scatter_kernel<<<cdiv(T, 256), 256, 0, stream>>>(Xc, tv, T, (float*)d_out);
}
